// DecoderLayer_62113817035144
// MI455X (gfx1250) — compile-verified, matrix operand order checked by reading, not run on hardware
//
#include <hip/hip_runtime.h>
#include <hip/hip_bf16.h>
#include <math.h>

typedef _Float16 half_t;
typedef __attribute__((ext_vector_type(16))) _Float16 v16h;
typedef __attribute__((ext_vector_type(8)))  _Float16 v8h;
typedef __attribute__((ext_vector_type(8)))  float    v8f;
typedef __attribute__((ext_vector_type(4)))  int      v4i;

#define EMBED 1024
#define NHEADS 16
#define HDIM 64
#define BATCH 4
#define SEQ 2048
#define BT (BATCH*SEQ)
#define FFND (4*EMBED)
#define LN_EPS 1e-5f

// CDNA5 async global->LDS direct copy (ASYNCcnt path), compile-safe guard.
#if defined(__AMDGCN__) && __has_builtin(__builtin_amdgcn_global_load_async_to_lds_b128)
#define HAVE_ASYNC_LDS 1
typedef __attribute__((address_space(1))) v4i gv4i;
typedef __attribute__((address_space(3))) v4i lv4i;
#else
#define HAVE_ASYNC_LDS 0
#endif

// CDNA5 WMMA 16-bit A/B fragment addressing (ISA 7.12.2):
// lane = M + 16*((K>>3)&1), element = (K&7) | ((K>>4)<<3)
__device__ __forceinline__ int frag_e(int k)          { return (k & 7) | ((k >> 4) << 3); }
__device__ __forceinline__ int frag_lane(int m, int k){ return m + (((k >> 3) & 1) << 4); }

__device__ __forceinline__ v8f wmma_f16(v16h a, v16h b, v8f c) {
  return __builtin_amdgcn_wmma_f32_16x16x32_f16(false, a, false, b, (short)0, c, false, false);
}

// 16-byte global -> LDS copy: async (ASYNCcnt path) when available.
__device__ __forceinline__ void copy16_g2l(const half_t* g, half_t* l) {
#if HAVE_ASYNC_LDS
  gv4i* gp = (gv4i*)(unsigned long long)g;
  lv4i* lp = (lv4i*)(unsigned int)(unsigned long long)l;
  __builtin_amdgcn_global_load_async_to_lds_b128(gp, lp, 0, 0);
#else
  *(v8h*)l = *(const v8h*)g;
#endif
}

__device__ __forceinline__ void wait_stage() {
#if HAVE_ASYNC_LDS
#if __has_builtin(__builtin_amdgcn_s_wait_asynccnt)
  __builtin_amdgcn_s_wait_asynccnt(0);
#else
  asm volatile("s_wait_asynccnt 0x0" ::: "memory");
#endif
#endif
}

// ---------------------------------------------------------------------------
// One-shot weight conversion f32 -> f16 with pre-swizzle into WMMA fragment
// order: frag id = (k/32)*(N/16) + (n/16); inside: lane*16 + e.
// Makes GEMM B staging a pure linear b128 copy.
// ---------------------------------------------------------------------------
__global__ void cvt_swizzle_w(const float* __restrict__ in, half_t* __restrict__ out,
                              int Kd, int Nd) {
  int total = Kd * Nd;
  for (int idx = blockIdx.x * 256 + threadIdx.x; idx < total; idx += gridDim.x * 256) {
    int k = idx / Nd, n = idx % Nd;
    size_t o = ((size_t)(k >> 5) * (Nd >> 4) + (n >> 4)) * 512
             + frag_lane(n & 15, k & 31) * 16 + frag_e(k & 31);
    out[o] = (half_t)in[idx];
  }
}

// Per-head QKV weights [H, C, HD] -> per-head fragment order
__global__ void cvt_swizzle_whead(const float* __restrict__ in, half_t* __restrict__ out) {
  int total = NHEADS * EMBED * HDIM;
  for (int idx = blockIdx.x * 256 + threadIdx.x; idx < total; idx += gridDim.x * 256) {
    int h = idx >> 16;                 // C*HD = 65536
    int r = idx & 65535;
    int k = r >> 6, n = r & 63;        // k in [0,1024), n in [0,64)
    size_t o = (size_t)h * (EMBED * HDIM)
             + ((size_t)(k >> 5) * 4 + (n >> 4)) * 512
             + frag_lane(n & 15, k & 31) * 16 + frag_e(k & 31);
    out[o] = (half_t)in[idx];
  }
}

// ---------------------------------------------------------------------------
// Fused LayerNorm (row of 1024) + cast to f16.
// ---------------------------------------------------------------------------
__global__ __launch_bounds__(256) void ln_cast(const float* __restrict__ x,
                                               const float* __restrict__ g,
                                               const float* __restrict__ be,
                                               half_t* __restrict__ out) {
  __shared__ float red0[8];
  __shared__ float red1[8];
  __shared__ float stats[2];
  int row = blockIdx.x, tid = threadIdx.x;
  const float* xr = x + (size_t)row * EMBED;
  float v[4]; float s = 0.f, s2 = 0.f;
#pragma unroll
  for (int i = 0; i < 4; ++i) { v[i] = xr[tid + i * 256]; s += v[i]; s2 += v[i] * v[i]; }
#pragma unroll
  for (int off = 16; off > 0; off >>= 1) { s += __shfl_down(s, off, 32); s2 += __shfl_down(s2, off, 32); }
  int wave = tid >> 5, lane = tid & 31;
  if (lane == 0) { red0[wave] = s; red1[wave] = s2; }
  __syncthreads();
  if (tid == 0) {
    float ts = 0.f, ts2 = 0.f;
#pragma unroll
    for (int w = 0; w < 8; ++w) { ts += red0[w]; ts2 += red1[w]; }
    float mu = ts * (1.f / EMBED);
    float var = ts2 * (1.f / EMBED) - mu * mu;
    stats[0] = mu; stats[1] = rsqrtf(var + LN_EPS);
  }
  __syncthreads();
  float mu = stats[0], rs = stats[1];
  half_t* orow = out + (size_t)row * EMBED;
#pragma unroll
  for (int i = 0; i < 4; ++i) {
    int c = tid + i * 256;
    orow[c] = (half_t)((v[i] - mu) * rs * g[c] + be[c]);
  }
}

// ---------------------------------------------------------------------------
// Shared GEMM building blocks. Block tile 256(M) x 64(N), K-step 32,
// double-buffered LDS, 8 waves, wave owns 32 rows (2 A-subtiles, 8 WMMA/step).
// ---------------------------------------------------------------------------
__device__ __forceinline__ void stage_a(const half_t* __restrict__ A, half_t* sA,
                                        int m0, int kt, int Kdim, int tid) {
#pragma unroll
  for (int c = 0; c < 4; ++c) {
    int chunk = tid + c * 256;
    int r = chunk >> 2, kc = (chunk & 3) * 8;
    const half_t* src = A + (size_t)(m0 + r) * Kdim + kt + kc;
    half_t* dst = sA + (r >> 4) * 512 + frag_lane(r & 15, kc) * 16 + frag_e(kc);
    copy16_g2l(src, dst);
  }
}

__device__ __forceinline__ void stage_b(const half_t* __restrict__ Bsw, half_t* sB,
                                        size_t fragBase, int tid) {
  copy16_g2l(Bsw + fragBase + (size_t)tid * 8, sB + tid * 8);
}

__device__ __forceinline__ void compute_step(const half_t* sA, const half_t* sB,
                                             int wave, int lane, v8f (&acc)[2][4]) {
  v16h a0 = *(const v16h*)(sA + (2 * wave) * 512 + lane * 16);
  v16h a1 = *(const v16h*)(sA + (2 * wave + 1) * 512 + lane * 16);
#pragma unroll
  for (int nt = 0; nt < 4; ++nt) {
    v16h b = *(const v16h*)(sB + nt * 512 + lane * 16);
    acc[0][nt] = wmma_f16(a0, b, acc[0][nt]);
    acc[1][nt] = wmma_f16(a1, b, acc[1][nt]);
  }
}

// ---------------------------------------------------------------------------
// Per-head projection GEMM: out[b,h,t,:] = Xn[b*T+t,:] @ W[h,:,:]
// ---------------------------------------------------------------------------
__global__ __launch_bounds__(256) void qkv_gemm(const half_t* __restrict__ Xn,
                                                const half_t* __restrict__ Wsw,
                                                half_t* __restrict__ Out) {
  __shared__ half_t sA[2][256 * 32];
  __shared__ half_t sB[2][32 * 64];
  int h = blockIdx.y;
  int m0 = blockIdx.x * 256;
  int tid = threadIdx.x, wave = tid >> 5, lane = tid & 31;
  const half_t* Bsw = Wsw + (size_t)h * (EMBED * HDIM);
  v8f acc[2][4] = {};
  stage_a(Xn, sA[0], m0, 0, EMBED, tid);
  stage_b(Bsw, sB[0], 0, tid);
  int cur = 0;
  for (int kt = 0; kt < EMBED; kt += 32) {
    wait_stage();
    __syncthreads();
    if (kt + 32 < EMBED) {
      stage_a(Xn, sA[cur ^ 1], m0, kt + 32, EMBED, tid);
      stage_b(Bsw, sB[cur ^ 1], (size_t)((kt + 32) >> 5) * 2048, tid);
    }
    compute_step(sA[cur], sB[cur], wave, lane, acc);
    cur ^= 1;
  }
  int nn = lane & 15, khf = lane >> 4;
#pragma unroll
  for (int sfr = 0; sfr < 2; ++sfr)
#pragma unroll
    for (int nt = 0; nt < 4; ++nt)
#pragma unroll
      for (int r = 0; r < 8; ++r) {
        int grow = m0 + wave * 32 + sfr * 16 + r + 8 * khf;
        int bb = grow >> 11, t = grow & (SEQ - 1);
        Out[(((size_t)bb * NHEADS + h) * SEQ + t) * HDIM + nt * 16 + nn] = (half_t)acc[sfr][nt][r];
      }
}

// ---------------------------------------------------------------------------
// Flash attention, causal.  Grid: (T/128, H, B), 8 waves/block.
// ---------------------------------------------------------------------------
__global__ __launch_bounds__(256) void attn_kernel(const half_t* __restrict__ Q,
                                                   const half_t* __restrict__ K,
                                                   const half_t* __restrict__ V,
                                                   half_t* __restrict__ O) {
  __shared__ half_t sP[8 * 512];   // per-wave 16x32 P tile in A-fragment layout
  int tid = threadIdx.x, wave = tid >> 5, lane = tid & 31;
  int h = blockIdx.y, b = blockIdx.z;
  size_t bh = (size_t)b * NHEADS + h;
  const half_t* Qb = Q + bh * SEQ * HDIM;
  const half_t* Kb = K + bh * SEQ * HDIM;
  const half_t* Vb = V + bh * SEQ * HDIM;
  int qBase = blockIdx.x * 128 + wave * 16;
  int m = lane & 15, khf = lane >> 4;

  const half_t* qrow = Qb + (size_t)(qBase + m) * HDIM;
  v16h qa[2];
#pragma unroll
  for (int i = 0; i < 2; ++i) {
    v8h lo = *(const v8h*)(qrow + i * 32 + khf * 8);
    v8h hi = *(const v8h*)(qrow + i * 32 + 16 + khf * 8);
#pragma unroll
    for (int e = 0; e < 8; ++e) { qa[i][e] = lo[e]; qa[i][e + 8] = hi[e]; }
  }

  v8f o[4] = {};
  float mi[8], li[8];
#pragma unroll
  for (int r = 0; r < 8; ++r) { mi[r] = -INFINITY; li[r] = 0.f; }
  half_t* pw = sP + wave * 512;

  int kEnd = qBase + 16;
  for (int kb = 0; kb < kEnd; kb += 32) {
    float sc[2][8], pnew[2][8], mnew[8];
#pragma unroll
    for (int r = 0; r < 8; ++r) mnew[r] = mi[r];

#pragma unroll
    for (int kt2 = 0; kt2 < 2; ++kt2) {
      int sBase = kb + kt2 * 16;
      const half_t* krow = Kb + (size_t)(sBase + m) * HDIM;
      v8f s = {};
#pragma unroll
      for (int i = 0; i < 2; ++i) {
        v8h lo = *(const v8h*)(krow + i * 32 + khf * 8);
        v8h hi = *(const v8h*)(krow + i * 32 + 16 + khf * 8);
        v16h kf;
#pragma unroll
        for (int e = 0; e < 8; ++e) { kf[e] = lo[e]; kf[e + 8] = hi[e]; }
        s = wmma_f16(qa[i], kf, s);
      }
      int key = sBase + (lane & 15);
#pragma unroll
      for (int r = 0; r < 8; ++r) {
        int qr = qBase + r + 8 * khf;
        float val = s[r] * 0.125f;
        if (key > qr) val = -INFINITY;
        sc[kt2][r] = val;
        mnew[r] = fmaxf(mnew[r], val);
      }
    }
#pragma unroll
    for (int r = 0; r < 8; ++r) {
      float v = mnew[r];
#pragma unroll
      for (int off = 8; off > 0; off >>= 1) v = fmaxf(v, __shfl_xor(v, off, 16));
      mnew[r] = v;
    }
#pragma unroll
    for (int r = 0; r < 8; ++r) {
      float fac = (mi[r] <= -1e37f) ? 0.f : __expf(mi[r] - mnew[r]);
      float psum = 0.f;
#pragma unroll
      for (int kt2 = 0; kt2 < 2; ++kt2) {
        float p = (sc[kt2][r] <= -1e37f) ? 0.f : __expf(sc[kt2][r] - mnew[r]);
        pnew[kt2][r] = p; psum += p;
      }
#pragma unroll
      for (int off = 8; off > 0; off >>= 1) psum += __shfl_xor(psum, off, 16);
      li[r] = li[r] * fac + psum;
      mi[r] = mnew[r];
#pragma unroll
      for (int nt = 0; nt < 4; ++nt) o[nt][r] *= fac;
    }
#pragma unroll
    for (int kt2 = 0; kt2 < 2; ++kt2)
#pragma unroll
      for (int r = 0; r < 8; ++r) {
        int M = r + 8 * khf;
        int N = kt2 * 16 + (lane & 15);
        pw[frag_lane(M, N) * 16 + frag_e(N)] = (half_t)pnew[kt2][r];
      }
    asm volatile("s_wait_dscnt 0" ::: "memory");
    v16h pa = *(const v16h*)(pw + lane * 16);
#pragma unroll
    for (int nt = 0; nt < 4; ++nt) {
      v16h vf;
#pragma unroll
      for (int e = 0; e < 16; ++e) {
        int kl = (e & 7) | (khf << 3) | ((e >> 3) << 4);
        vf[e] = Vb[(size_t)(kb + kl) * HDIM + nt * 16 + (lane & 15)];
      }
      o[nt] = wmma_f16(pa, vf, o[nt]);
    }
  }
#pragma unroll
  for (int nt = 0; nt < 4; ++nt)
#pragma unroll
    for (int r = 0; r < 8; ++r) {
      int qr = qBase + r + 8 * khf;
      float inv = 1.f / li[r];
      O[(((size_t)b * SEQ + qr) * NHEADS + h) * HDIM + nt * 16 + (lane & 15)] = (half_t)(o[nt][r] * inv);
    }
}

// ---------------------------------------------------------------------------
// Generic epilogue GEMM: out = A(f16)[M,K] @ Bsw(f16 swizzled)[K,N] + bias
//   do_gelu=1: outH = gelu_exact(acc+bias) as f16
//   do_gelu=0: outF = acc + bias + (resid ? resid : 0) as f32
// ---------------------------------------------------------------------------
__global__ __launch_bounds__(256) void gemm_ep(const half_t* __restrict__ A,
                                               const half_t* __restrict__ Bsw,
                                               const float* __restrict__ bias,
                                               const float* __restrict__ resid,
                                               float* __restrict__ outF,
                                               half_t* __restrict__ outH,
                                               int Ndim, int Kdim, int do_gelu) {
  __shared__ half_t sA[2][256 * 32];
  __shared__ half_t sB[2][32 * 64];
  int m0 = blockIdx.x * 256, n0 = blockIdx.y * 64;
  int tid = threadIdx.x, wave = tid >> 5, lane = tid & 31;
  v8f acc[2][4] = {};
  size_t fragStride = (size_t)(Ndim >> 4) * 512;
  size_t fb0 = (size_t)(n0 >> 4) * 512;
  stage_a(A, sA[0], m0, 0, Kdim, tid);
  stage_b(Bsw, sB[0], fb0, tid);
  int cur = 0;
  for (int kt = 0; kt < Kdim; kt += 32) {
    wait_stage();
    __syncthreads();
    if (kt + 32 < Kdim) {
      stage_a(A, sA[cur ^ 1], m0, kt + 32, Kdim, tid);
      stage_b(Bsw, sB[cur ^ 1], fb0 + (size_t)((kt + 32) >> 5) * fragStride, tid);
    }
    compute_step(sA[cur], sB[cur], wave, lane, acc);
    cur ^= 1;
  }
  int nn = lane & 15, khf = lane >> 4;
#pragma unroll
  for (int sfr = 0; sfr < 2; ++sfr)
#pragma unroll
    for (int nt = 0; nt < 4; ++nt)
#pragma unroll
      for (int r = 0; r < 8; ++r) {
        int row = m0 + wave * 32 + sfr * 16 + r + 8 * khf;
        int col = n0 + nt * 16 + nn;
        float v = acc[sfr][nt][r] + bias[col];
        size_t idx = (size_t)row * Ndim + col;
        if (do_gelu) {
          outH[idx] = (half_t)(0.5f * v * (1.f + erff(v * 0.70710678118f)));
        } else {
          outF[idx] = v + (resid ? resid[idx] : 0.f);
        }
      }
}

// ---------------------------------------------------------------------------
extern "C" void kernel_launch(void* const* d_in, const int* in_sizes, int n_in,
                              void* d_out, int out_size, void* d_ws, size_t ws_size,
                              hipStream_t stream) {
  (void)in_sizes; (void)n_in; (void)out_size; (void)ws_size;
  const float* x    = (const float*)d_in[0];
  const float* Wq   = (const float*)d_in[1];
  const float* Wk   = (const float*)d_in[2];
  const float* Wv   = (const float*)d_in[3];
  const float* Wo   = (const float*)d_in[4];
  const float* bo   = (const float*)d_in[5];
  const float* g1   = (const float*)d_in[6];
  const float* be1  = (const float*)d_in[7];
  const float* g2   = (const float*)d_in[8];
  const float* be2  = (const float*)d_in[9];
  const float* W1   = (const float*)d_in[10];
  const float* bff1 = (const float*)d_in[11];
  const float* W2   = (const float*)d_in[12];
  const float* bff2 = (const float*)d_in[13];

  char* ws = (char*)d_ws;
  size_t off = 0;
  auto alloc = [&](size_t bytes) -> void* {
    void* p = ws + off; off += (bytes + 255) & ~(size_t)255; return p;
  };
  half_t* xn1   = (half_t*)alloc((size_t)BT * EMBED * 2);
  half_t* wq16  = (half_t*)alloc((size_t)NHEADS * EMBED * HDIM * 2);
  half_t* wk16  = (half_t*)alloc((size_t)NHEADS * EMBED * HDIM * 2);
  half_t* wv16  = (half_t*)alloc((size_t)NHEADS * EMBED * HDIM * 2);
  half_t* wo16  = (half_t*)alloc((size_t)EMBED * EMBED * 2);
  half_t* w116  = (half_t*)alloc((size_t)EMBED * FFND * 2);
  half_t* w216  = (half_t*)alloc((size_t)FFND * EMBED * 2);
  half_t* q16   = (half_t*)alloc((size_t)BT * EMBED * 2);
  half_t* k16   = (half_t*)alloc((size_t)BT * EMBED * 2);
  half_t* v16   = (half_t*)alloc((size_t)BT * EMBED * 2);
  half_t* at16  = (half_t*)alloc((size_t)BT * EMBED * 2);
  float*  res32 = (float*)alloc((size_t)BT * EMBED * 4);
  half_t* xn2   = (half_t*)alloc((size_t)BT * EMBED * 2);
  half_t* hid16 = (half_t*)alloc((size_t)BT * FFND * 2);

  cvt_swizzle_whead<<<1024, 256, 0, stream>>>(Wq, wq16);
  cvt_swizzle_whead<<<1024, 256, 0, stream>>>(Wk, wk16);
  cvt_swizzle_whead<<<1024, 256, 0, stream>>>(Wv, wv16);
  cvt_swizzle_w<<<1024, 256, 0, stream>>>(Wo, wo16, EMBED, EMBED);
  cvt_swizzle_w<<<2048, 256, 0, stream>>>(W1, w116, EMBED, FFND);
  cvt_swizzle_w<<<2048, 256, 0, stream>>>(W2, w216, FFND, EMBED);

  ln_cast<<<BT, 256, 0, stream>>>(x, g1, be1, xn1);

  dim3 gq(BT / 256, NHEADS);
  qkv_gemm<<<gq, 256, 0, stream>>>(xn1, wq16, q16);
  qkv_gemm<<<gq, 256, 0, stream>>>(xn1, wk16, k16);
  qkv_gemm<<<gq, 256, 0, stream>>>(xn1, wv16, v16);

  dim3 ga(SEQ / 128, NHEADS, BATCH);
  attn_kernel<<<ga, 256, 0, stream>>>(q16, k16, v16, at16);

  dim3 gp(BT / 256, EMBED / 64);
  gemm_ep<<<gp, 256, 0, stream>>>(at16, wo16, bo, x, res32, nullptr, EMBED, EMBED, 0);

  ln_cast<<<BT, 256, 0, stream>>>(res32, g2, be2, xn2);

  dim3 gf1(BT / 256, FFND / 64);
  gemm_ep<<<gf1, 256, 0, stream>>>(xn2, w116, bff1, nullptr, nullptr, hid16, FFND, EMBED, 1);

  dim3 gf2(BT / 256, EMBED / 64);
  gemm_ep<<<gf2, 256, 0, stream>>>(hid16, w216, bff2, res32, (float*)d_out, nullptr, EMBED, FFND, 0);
}
